// RoPEAttention_69226282877570
// MI455X (gfx1250) — matrix operand order repaired and checked by reading, not compiled, not run on hardware
//
#include <hip/hip_runtime.h>
#include <hip/hip_bf16.h>

typedef __attribute__((ext_vector_type(16))) __bf16 v16bf;
typedef __attribute__((ext_vector_type(8)))  float  v8f;
typedef __attribute__((ext_vector_type(8)))  __bf16 bf16x8;

// ---------------------------------------------------------------------------
// WMMA helper: D = A(16x32 bf16) * B(32x16 bf16) + C(16x16 f32)
// ---------------------------------------------------------------------------
__device__ __forceinline__ v8f wmma_bf16(v16bf a, v16bf b, v8f c) {
  return __builtin_amdgcn_wmma_f32_16x16x32_bf16(
      /*neg_a=*/false, a, /*neg_b=*/false, b,
      /*c_mod=*/(short)0, c, /*reuse_a=*/false, /*reuse_b=*/false);
}

// A fragment (16x32), source row-major [row][k], ld in elements.
// Lane hl (=lane&15) holds row hl; elements i: K = hi*8+i, 8+i: K = 16+hi*8+i.
__device__ __forceinline__ v16bf load_frag_a(const __bf16* base, int ld, int lane) {
  const int hl = lane & 15, hi = lane >> 4;
  const __bf16* p = base + (size_t)hl * ld + hi * 8;
  bf16x8 lo = *(const bf16x8*)(p);
  bf16x8 hi8 = *(const bf16x8*)(p + 16);
  v16bf f;
#pragma unroll
  for (int i = 0; i < 8; ++i) { f[i] = lo[i]; f[8 + i] = hi8[i]; }
  return f;
}

// B fragment (32x16), source stored [n][k] row-major (K-contiguous), ld in elems.
// Lane hl holds column hl; elements i: K = hi*8+i, 8+i: K = 16+hi*8+i.
__device__ __forceinline__ v16bf load_frag_b(const __bf16* base, int ld, int lane) {
  const int hl = lane & 15, hi = lane >> 4;
  const __bf16* p = base + (size_t)hl * ld + hi * 8;
  bf16x8 a = *(const bf16x8*)(p);
  bf16x8 b = *(const bf16x8*)(p + 16);
  v16bf f;
#pragma unroll
  for (int i = 0; i < 8; ++i) { f[i] = a[i]; f[8 + i] = b[i]; }
  return f;
}

// ---------------------------------------------------------------------------
// fp32 -> bf16 conversion
// ---------------------------------------------------------------------------
__global__ void cvt_bf16_kernel(const float* __restrict__ in,
                                __bf16* __restrict__ out, int n) {
  int i = blockIdx.x * blockDim.x + threadIdx.x;
  if (i < n) out[i] = (__bf16)in[i];
}

// ---------------------------------------------------------------------------
// Generic NT GEMM: C[M,N] = A[M,K] * B[N,K]^T  (A,B bf16, accum f32)
// Wave tile 16x64 (4 accums), 8 waves/block -> block tile 128x64.
// Ping-pong register buffering, k-step 64 per iteration: loads always land in
// the same registers (no rotation copies) and are issued one 32-wide k-tile
// ahead of the WMMAs that consume them.
// mode 0: write f32 C row-major [M][N]
// mode 1: QKV scatter: col e -> (which,h,dh); row m -> (b,t)
//         q,k -> [B,H,T,64] bf16 ; v -> transposed [B,H,64,T] bf16
// ---------------------------------------------------------------------------
__global__ __launch_bounds__(256) void gemm_bf16_kernel(
    const __bf16* __restrict__ A, const __bf16* __restrict__ Bm,
    int M, int N, int K, int mode,
    float* __restrict__ Cf,
    __bf16* __restrict__ Qo, __bf16* __restrict__ Ko, __bf16* __restrict__ Vt,
    int T, int H) {
  const int lane = threadIdx.x & 31;
  const int wave = threadIdx.x >> 5;
  const int mBase = blockIdx.y * 128 + wave * 16;
  const int nBase = blockIdx.x * 64;

  const __bf16* Arow = A + (size_t)mBase * K;

  v8f acc[4];
#pragma unroll
  for (int j = 0; j < 4; ++j)
#pragma unroll
    for (int r = 0; r < 8; ++r) acc[j][r] = 0.0f;

  // Prologue: buffer A <- k = 0
  v16bf aA = load_frag_a(Arow, K, lane);
  v16bf bA[4];
#pragma unroll
  for (int j = 0; j < 4; ++j)
    bA[j] = load_frag_b(Bm + (size_t)(nBase + j * 16) * K, K, lane);

  // K is a multiple of 64 for all call sites.
  for (int k0 = 0; k0 < K; k0 += 64) {
    // Buffer B <- k0 + 32 (always in range)
    v16bf aB = load_frag_a(Arow + k0 + 32, K, lane);
    v16bf bB[4];
#pragma unroll
    for (int j = 0; j < 4; ++j)
      bB[j] = load_frag_b(Bm + (size_t)(nBase + j * 16) * K + k0 + 32, K, lane);

    // speculative prefetch ahead on the A stream (global_prefetch_b8)
    __builtin_prefetch(Arow + k0 + 512, 0, 1);

    // Compute with buffer A (k0)
#pragma unroll
    for (int j = 0; j < 4; ++j) acc[j] = wmma_bf16(aA, bA[j], acc[j]);

    // Buffer A <- k0 + 64 (branchless wrap on final iteration: harmless reload)
    const int kn = (k0 + 64 < K) ? (k0 + 64) : 0;
    aA = load_frag_a(Arow + kn, K, lane);
#pragma unroll
    for (int j = 0; j < 4; ++j)
      bA[j] = load_frag_b(Bm + (size_t)(nBase + j * 16) * K + kn, K, lane);

    // Compute with buffer B (k0 + 32)
#pragma unroll
    for (int j = 0; j < 4; ++j) acc[j] = wmma_bf16(aB, bB[j], acc[j]);
  }

  const int hl = lane & 15, hi = lane >> 4;
#pragma unroll
  for (int j = 0; j < 4; ++j) {
#pragma unroll
    for (int r = 0; r < 8; ++r) {
      const int row = mBase + r + 8 * hi;
      const int col = nBase + j * 16 + hl;
      const float v = acc[j][r];
      if (mode == 0) {
        Cf[(size_t)row * N + col] = v;
      } else {
        const int which = col >> 10;     // 0=q 1=k 2=v  (D = 1024)
        const int rc = col & 1023;
        const int h = rc >> 6;           // Dh = 64
        const int dh = rc & 63;
        const int b = row / T;
        const int t = row - b * T;
        const __bf16 bv = (__bf16)v;
        if (which == 2) {
          Vt[((size_t)(b * H + h) * 64 + dh) * T + t] = bv;
        } else {
          const size_t idx = ((size_t)(b * H + h) * T + t) * 64 + dh;
          if (which == 0) Qo[idx] = bv; else Ko[idx] = bv;
        }
      }
    }
  }
}

// ---------------------------------------------------------------------------
// RoPE on q and k, layout [B*H][T][64] bf16, interleaved pairs (even,odd).
// theta_p = 10000^(-2p/64), ang = t * theta_p.
// ---------------------------------------------------------------------------
__global__ void rope_kernel(__bf16* __restrict__ Q, __bf16* __restrict__ K,
                            int T, int total_pairs) {
  int idx = blockIdx.x * blockDim.x + threadIdx.x;
  if (idx >= total_pairs) return;
  const int p = idx & 31;            // pair index within head dim
  const int t = (idx >> 5) % T;      // sequence position
  // ln(10000)/32
  const float theta = __expf((float)p * (-0.28782313662425572f));
  const float ang = (float)t * theta;
  float s, c;
  __sincosf(ang, &s, &c);
  const size_t base = (size_t)idx * 2;
  {
    float e = (float)Q[base], o = (float)Q[base + 1];
    Q[base]     = (__bf16)(e * c - o * s);
    Q[base + 1] = (__bf16)(e * s + o * c);
  }
  {
    float e = (float)K[base], o = (float)K[base + 1];
    K[base]     = (__bf16)(e * c - o * s);
    K[base + 1] = (__bf16)(e * s + o * c);
  }
}

// ---------------------------------------------------------------------------
// Flash attention per (b,h). Block = 4 waves; each wave owns 16 query rows.
// Q,K: [BH][T][64] bf16, Vt: [BH][64][T] bf16, Out: [B*T][H*64] bf16.
// 64 keys per loop iteration with ping-pong K-fragment buffers (no rotation
// copies); the other tile's K loads overlap each tile's softmax VALU work.
// ---------------------------------------------------------------------------
__global__ __launch_bounds__(128) void attn_kernel(
    const __bf16* __restrict__ Q, const __bf16* __restrict__ Kc,
    const __bf16* __restrict__ Vt, __bf16* __restrict__ Out,
    int T, int H) {
  __shared__ __attribute__((aligned(16))) __bf16 Plds[4][16 * 32];

  const int lane = threadIdx.x & 31;
  const int wave = threadIdx.x >> 5;
  const int bh = blockIdx.x;                  // b*H + h
  const int rowBase = blockIdx.y * 64 + wave * 16;
  const int hl = lane & 15, hi = lane >> 4;

  const __bf16* q  = Q  + (size_t)bh * T * 64;
  const __bf16* k  = Kc + (size_t)bh * T * 64;
  const __bf16* vt = Vt + (size_t)bh * 64 * T;   // [64][T]

  // Q fragments for this wave's 16 rows (Dh=64 -> two 32-wide K steps)
  const v16bf qa0 = load_frag_a(q + (size_t)rowBase * 64 + 0, 64, lane);
  const v16bf qa1 = load_frag_a(q + (size_t)rowBase * 64 + 32, 64, lane);

  float m_r[8], l_r[8], alpha[8];
#pragma unroll
  for (int r = 0; r < 8; ++r) { m_r[r] = -INFINITY; l_r[r] = 0.0f; }
  v8f o[4];
#pragma unroll
  for (int j = 0; j < 4; ++j)
#pragma unroll
    for (int r = 0; r < 8; ++r) o[j][r] = 0.0f;

  const float scale = 0.125f;  // 1/sqrt(64)
  __bf16* pl = &Plds[wave][0];

  // Process one 32-key tile given its 4 preloaded K fragments.
  auto process_tile = [&](int j0, const v16bf& kb00, const v16bf& kb01,
                          const v16bf& kb10, const v16bf& kb11) {
    // S = Q * K^T for 32 keys (two 16-key subtiles)
    v8f s0, s1;
#pragma unroll
    for (int r = 0; r < 8; ++r) { s0[r] = 0.0f; s1[r] = 0.0f; }
    s0 = wmma_bf16(qa0, kb00, s0);
    s0 = wmma_bf16(qa1, kb01, s0);
    s1 = wmma_bf16(qa0, kb10, s1);
    s1 = wmma_bf16(qa1, kb11, s1);

    // Issue this tile's V loads early (overlap softmax VALU work)
    v16bf vb[4];
#pragma unroll
    for (int j = 0; j < 4; ++j)
      vb[j] = load_frag_b(vt + (size_t)(j * 16) * T + j0, T, lane);

    // Online softmax: row r lives on lanes [hi*16 .. hi*16+15], one col/lane.
#pragma unroll
    for (int r = 0; r < 8; ++r) {
      float a0 = s0[r] * scale, a1 = s1[r] * scale;
      float mx = fmaxf(a0, a1);
#pragma unroll
      for (int d = 8; d >= 1; d >>= 1) mx = fmaxf(mx, __shfl_xor(mx, d, 32));
      const float mn = fmaxf(m_r[r], mx);
      alpha[r] = __expf(m_r[r] - mn);
      const float p0 = __expf(a0 - mn), p1 = __expf(a1 - mn);
      s0[r] = p0; s1[r] = p1;
      float sum = p0 + p1;
#pragma unroll
      for (int d = 8; d >= 1; d >>= 1) sum += __shfl_xor(sum, d, 32);
      l_r[r] = l_r[r] * alpha[r] + sum;
      m_r[r] = mn;
    }

    // Rescale running output
#pragma unroll
    for (int j = 0; j < 4; ++j)
#pragma unroll
      for (int r = 0; r < 8; ++r) o[j][r] *= alpha[r];

    // Stage P (C layout) -> LDS -> A-fragment layout
#pragma unroll
    for (int r = 0; r < 8; ++r) {
      pl[(r + 8 * hi) * 32 + hl]      = (__bf16)s0[r];
      pl[(r + 8 * hi) * 32 + 16 + hl] = (__bf16)s1[r];
    }
    __syncthreads();
    const v16bf pf = load_frag_a(pl, 32, lane);

    // O += P(16x32) * V(32x64)
#pragma unroll
    for (int j = 0; j < 4; ++j) o[j] = wmma_bf16(pf, vb[j], o[j]);
    __syncthreads();
  };

  // K-fragment loader for a 16-key subtile pair starting at key j.
  // Buffer A <- keys [0, 32)
  v16bf ka00 = load_frag_b(k + 0, 64, lane);
  v16bf ka01 = load_frag_b(k + 32, 64, lane);
  v16bf ka10 = load_frag_b(k + (size_t)16 * 64 + 0, 64, lane);
  v16bf ka11 = load_frag_b(k + (size_t)16 * 64 + 32, 64, lane);

  // T is a multiple of 64.
  for (int j0 = 0; j0 < T; j0 += 64) {
    // Buffer B <- keys [j0+32, j0+64)
    const size_t o1 = (size_t)(j0 + 32) * 64;
    v16bf kb00 = load_frag_b(k + o1, 64, lane);
    v16bf kb01 = load_frag_b(k + o1 + 32, 64, lane);
    v16bf kb10 = load_frag_b(k + o1 + (size_t)16 * 64, 64, lane);
    v16bf kb11 = load_frag_b(k + o1 + (size_t)16 * 64 + 32, 64, lane);

    process_tile(j0, ka00, ka01, ka10, ka11);

    // Buffer A <- keys [j0+64, j0+96) (branchless wrap on final iteration)
    const int jn = (j0 + 64 < T) ? (j0 + 64) : 0;
    const size_t o2 = (size_t)jn * 64;
    ka00 = load_frag_b(k + o2, 64, lane);
    ka01 = load_frag_b(k + o2 + 32, 64, lane);
    ka10 = load_frag_b(k + o2 + (size_t)16 * 64, 64, lane);
    ka11 = load_frag_b(k + o2 + (size_t)16 * 64 + 32, 64, lane);

    process_tile(j0 + 32, kb00, kb01, kb10, kb11);
  }

  // Normalize and write Out[b*T + row][h*64 + dh] (bf16)
  const int b = bh / H, h = bh - b * H;
#pragma unroll
  for (int j = 0; j < 4; ++j) {
#pragma unroll
    for (int r = 0; r < 8; ++r) {
      const int row = rowBase + r + 8 * hi;
      const int dh = j * 16 + hl;
      const float val = o[j][r] / l_r[r];
      Out[(size_t)(b * T + row) * (H * 64) + h * 64 + dh] = (__bf16)val;
    }
  }
}

// ---------------------------------------------------------------------------
// Host side
// ---------------------------------------------------------------------------
extern "C" void kernel_launch(void* const* d_in, const int* in_sizes, int n_in,
                              void* d_out, int out_size, void* d_ws, size_t ws_size,
                              hipStream_t stream) {
  (void)in_sizes; (void)n_in; (void)out_size; (void)ws_size;
  const int B = 4, T = 2048, D = 1024, H = 16;
  const int M = B * T;            // 8192
  const int N_QKV = 3 * D;        // 3072
  const int K = D;                // 1024

  const float* x     = (const float*)d_in[0];
  const float* Wqkv  = (const float*)d_in[1];
  const float* Wproj = (const float*)d_in[2];
  float* out = (float*)d_out;

  char* w = (char*)d_ws;
  __bf16* xb    = (__bf16*)(w);                         // 16 MB
  __bf16* wqkvb = (__bf16*)(w + 16777216);              // 6 MB
  __bf16* wprjb = (__bf16*)(w + 23068672);              // 2 MB
  __bf16* qb    = (__bf16*)(w + 25165824);              // 16 MB [B,H,T,64]
  __bf16* kb    = (__bf16*)(w + 41943040);              // 16 MB [B,H,T,64]
  __bf16* vtb   = (__bf16*)(w + 58720256);              // 16 MB [B,H,64,T]
  __bf16* ab    = (__bf16*)(w + 75497472);              // 16 MB [B*T, D]

  // 1) fp32 -> bf16 conversions
  {
    int n = M * K;                              // 8,388,608
    cvt_bf16_kernel<<<(n + 255) / 256, 256, 0, stream>>>(x, xb, n);
    n = N_QKV * K;                              // 3,145,728
    cvt_bf16_kernel<<<(n + 255) / 256, 256, 0, stream>>>(Wqkv, wqkvb, n);
    n = D * K;                                  // 1,048,576
    cvt_bf16_kernel<<<(n + 255) / 256, 256, 0, stream>>>(Wproj, wprjb, n);
  }

  // 2) QKV projection GEMM + scatter (mode 1)
  {
    dim3 grid(N_QKV / 64, M / 128);             // 48 x 64
    gemm_bf16_kernel<<<grid, 256, 0, stream>>>(xb, wqkvb, M, N_QKV, K,
                                               /*mode=*/1, nullptr,
                                               qb, kb, vtb, T, H);
  }

  // 3) RoPE on q and k
  {
    const int total_pairs = B * H * T * 32;     // 4,194,304
    rope_kernel<<<(total_pairs + 255) / 256, 256, 0, stream>>>(qb, kb, T, total_pairs);
  }

  // 4) Flash attention
  {
    dim3 grid(B * H, T / 64);                   // 64 x 32
    attn_kernel<<<grid, 128, 0, stream>>>(qb, kb, vtb, ab, T, H);
  }

  // 5) Output projection GEMM (mode 0, f32 out)
  {
    dim3 grid(D / 64, M / 128);                 // 16 x 64
    gemm_bf16_kernel<<<grid, 256, 0, stream>>>(ab, wprjb, M, D, K,
                                               /*mode=*/0, out,
                                               nullptr, nullptr, nullptr, T, H);
  }
}
